// PipelinedMoEBlock_15453292331376
// MI455X (gfx1250) — matrix operand order repaired and checked by reading, not compiled, not run on hardware
//
#include <hip/hip_runtime.h>
#include <math.h>

// ---------------------------------------------------------------------------
// MI455X / gfx1250 implementation of the attention + top2-MoE block.
// All matmuls use v_wmma_f32_16x16x32_bf16 (bf16 in, f32 accumulate).
// GEMM A-tiles staged with global_load_async_to_lds_b128 (ASYNCcnt path);
// weight stream prefetched with global_prefetch_b8.
// ---------------------------------------------------------------------------

typedef __bf16 bf16_t;
typedef __bf16 v16bf __attribute__((ext_vector_type(16)));
typedef float  v8f   __attribute__((ext_vector_type(8)));
typedef __attribute__((address_space(3))) char lds_char;

#define T_TOK 2048   // B*S
#define DMODEL 1024
#define NHEAD 16
#define HDIM 64
#define NEXP 8
#define FFDIM 2048
#define SEQ 512
#define NB 4

__device__ inline float gelu_f(float x) {
    const float c = 0.7978845608028654f;  // sqrt(2/pi)
    return 0.5f * x * (1.f + tanhf(c * (x + 0.044715f * x * x * x)));
}

// ---------------------------------------------------------------------------
// LayerNorm: fp32 in -> bf16 out (optionally copy input fp32 to `cpy`)
// ---------------------------------------------------------------------------
__global__ __launch_bounds__(256)
void ln_kernel(const float* __restrict__ x, const float* __restrict__ g,
               const float* __restrict__ b, bf16_t* __restrict__ out,
               float* __restrict__ cpy) {
    const int D = DMODEL;
    int tok = blockIdx.x;
    const float* xr = x + (size_t)tok * D;
    __shared__ float sa[256], sb[256];
    float a = 0.f, q = 0.f;
    for (int i = threadIdx.x; i < D; i += 256) { float v = xr[i]; a += v; q += v * v; }
    sa[threadIdx.x] = a; sb[threadIdx.x] = q;
    __syncthreads();
    for (int st = 128; st > 0; st >>= 1) {
        if ((int)threadIdx.x < st) { sa[threadIdx.x] += sa[threadIdx.x + st]; sb[threadIdx.x] += sb[threadIdx.x + st]; }
        __syncthreads();
    }
    float mu = sa[0] * (1.f / D);
    float var = sb[0] * (1.f / D) - mu * mu;
    float rs = rsqrtf(var + 1e-5f);
    for (int i = threadIdx.x; i < D; i += 256) {
        float v = xr[i];
        out[(size_t)tok * D + i] = (bf16_t)((v - mu) * rs * g[i] + b[i]);
        if (cpy) cpy[(size_t)tok * D + i] = v;
    }
}

// ---------------------------------------------------------------------------
// Generic tiled WMMA GEMM: C[M,N] = epilogue(A[M,K](bf16) @ B[K,N](fp32->bf16))
//   EPI 0: store bf16              (qkv projection)
//   EPI 1: store fp32 acc+addX     (Wo projection + residual)
//   EPI 2: store bf16 gelu(acc)    (MoE up-proj)
//   EPI 3: outF[rowIdx[i]] += rowW[i]*acc  (MoE down-proj scatter-accumulate)
// GATHER_A: A row i comes from rowIdx[i] (MoE up-proj token gather).
// Block: 256 thr = 8 waves; tile 64(M) x 128(N); wave = 16x64: one A frag
// reused across 4 WMMAs per K-chunk. A tile staged via async global->LDS.
// ---------------------------------------------------------------------------
template <int EPI, bool GATHER_A>
__global__ __launch_bounds__(256)
void gemm_bf16(const bf16_t* __restrict__ A, const float* __restrict__ Bw,
               float* __restrict__ outF, bf16_t* __restrict__ outH,
               const float* __restrict__ addX,
               const int* __restrict__ rowIdx, const float* __restrict__ rowW,
               const int* __restrict__ rowCount,
               int M, int N, int K) {
    __shared__ bf16_t As[64][34];
    __shared__ bf16_t Bs[32][130];

    int cnt = rowCount ? rowCount[0] : M;
    int m0 = blockIdx.y * 64;
    int n0 = blockIdx.x * 128;
    if (m0 >= cnt) return;

    int tid = threadIdx.x;
    int lane = tid & 31;
    int wid = tid >> 5;
    int wm = (wid & 3) * 16;   // wave row offset in tile
    int wn = (wid >> 2) * 64;  // wave col offset in tile

    v8f acc[4] = {{}, {}, {}, {}};

    // A-tile staging addresses (64x32 bf16 = one b128 per thread per chunk)
    int ar = tid >> 2, aq4 = tid & 3;
    int agr = m0 + ar;
    int agrow;
    if (GATHER_A) {
        int ii = agr < cnt ? agr : (cnt - 1);
        agrow = rowIdx[ii];
    } else {
        agrow = agr;
    }
    const bf16_t* abase = A + (size_t)agrow * K + aq4 * 8;
    unsigned aldsoff = (unsigned)(unsigned long long)(lds_char*)(&As[ar][aq4 * 8]);

    for (int k0 = 0; k0 < K; k0 += 32) {
        // ---- stage A tile: async global -> LDS, 16B per lane (ASYNCcnt) ----
        {
            unsigned long long ga = (unsigned long long)(abase + k0);
            asm volatile("global_load_async_to_lds_b128 %0, %1, off"
                         :: "v"(aldsoff), "v"(ga) : "memory");
        }
        // ---- stage B tile (32x128), fp32 -> bf16 on the way into LDS ----
#pragma unroll
        for (int i = 0; i < 4; ++i) {
            int e = tid + i * 256;
            int r = e >> 5, c4 = e & 31;
            const float* srcp = Bw + (size_t)(k0 + r) * N + n0 + c4 * 4;
            float4 d = *reinterpret_cast<const float4*>(srcp);
            Bs[r][c4 * 4 + 0] = (bf16_t)d.x;
            Bs[r][c4 * 4 + 1] = (bf16_t)d.y;
            Bs[r][c4 * 4 + 2] = (bf16_t)d.z;
            Bs[r][c4 * 4 + 3] = (bf16_t)d.w;
            if (i == 0 && k0 + 32 < K)
                __builtin_prefetch(srcp + (size_t)32 * N, 0, 0);  // next K-chunk of weights
        }
        asm volatile("s_wait_asynccnt 0x0" ::: "memory");
        __syncthreads();

        // ---- fragments: A and B both use contiguous k = 16*(lane>>4) + j ----
        v16bf af;
        int arow = wm + (lane & 15);
        int kb = (lane >> 4) * 16;
#pragma unroll
        for (int j = 0; j < 16; ++j) {
            int kk = kb + j;
            af[j] = As[arow][kk];
        }
        int bcol = lane & 15;
        int kbb = (lane >> 4) * 16;
#pragma unroll
        for (int t = 0; t < 4; ++t) {
            v16bf bfg;
#pragma unroll
            for (int j = 0; j < 16; ++j) bfg[j] = Bs[kbb + j][wn + t * 16 + bcol];
            acc[t] = __builtin_amdgcn_wmma_f32_16x16x32_bf16(false, af, false, bfg, (short)0, acc[t], false, false);
        }
        __syncthreads();
    }

    // ---- epilogue: C layout m = r + 8*(lane>>4), n = lane&15 ----
    int hi8 = (lane & 16) ? 8 : 0;
    int ncol = lane & 15;
#pragma unroll
    for (int r = 0; r < 8; ++r) {
        int i = m0 + wm + r + hi8;
        if (i >= cnt) continue;
#pragma unroll
        for (int t = 0; t < 4; ++t) {
            float v = acc[t][r];
            int c = n0 + wn + t * 16 + ncol;
            size_t o = (size_t)i * N + c;
            if (EPI == 0) {
                outH[o] = (bf16_t)v;
            } else if (EPI == 1) {
                outF[o] = v + addX[o];
            } else if (EPI == 2) {
                outH[o] = (bf16_t)gelu_f(v);
            } else {
                int gr = rowIdx[i];
                float w = rowW[i];
                outF[(size_t)gr * N + c] += w * v;
            }
        }
    }
}

// ---------------------------------------------------------------------------
// Causal flash attention, all WMMA. grid = (S/128, H, B), 256 thr = 8 waves.
// Each wave owns 16 q rows x full HD=64. Key chunks of 32; per chunk:
//   4 WMMA for scores (2 key halves x 2 hd halves), 4 WMMA for P@V.
// qkv layout: [token][3*D] with col = which*D + h*64 + hd.
// ---------------------------------------------------------------------------
__global__ __launch_bounds__(256)
void attn_kernel(const bf16_t* __restrict__ qkv, bf16_t* __restrict__ o) {
    __shared__ bf16_t P[8][16][34];

    int qb = blockIdx.x;   // 0..3
    int h  = blockIdx.y;
    int b  = blockIdx.z;
    int lane = threadIdx.x & 31;
    int wid  = threadIdx.x >> 5;
    int qbase = qb * 128 + wid * 16;

    const int S = SEQ, D3 = 3 * DMODEL;
    int m = lane & 15;
    int kb = (lane >> 4) * 16;

    // Q fragments (hd 0..31, 32..63)
    v16bf qf[2];
    {
        size_t qoff = (size_t)(b * S + qbase + m) * D3 + h * HDIM;
#pragma unroll
        for (int c = 0; c < 2; ++c)
#pragma unroll
            for (int j = 0; j < 16; ++j) {
                int kk = kb + j;
                qf[c][j] = qkv[qoff + c * 32 + kk];
            }
    }

    v8f of[4] = {{}, {}, {}, {}};
    float mst[8], lst[8];
#pragma unroll
    for (int r = 0; r < 8; ++r) { mst[r] = -1e30f; lst[r] = 0.f; }

    int hi8 = (lane & 16) ? 8 : 0;
    int nch = qb * 4 + 4;  // uniform over block: covers keys up to block's last q row

    for (int ch = 0; ch < nch; ++ch) {
        int kc = ch * 32;
        // ---- scores: two 16x16 tiles over keys [kc,kc+16) and [kc+16,kc+32) ----
        v8f s0 = {}, s1 = {};
        {
            int keyn = lane & 15;
            int kk16 = (lane >> 4) * 16;
#pragma unroll
            for (int half = 0; half < 2; ++half) {
#pragma unroll
                for (int c = 0; c < 2; ++c) {
                    v16bf kf;
#pragma unroll
                    for (int j = 0; j < 16; ++j) {
                        int key = kc + half * 16 + keyn;
                        int hd = c * 32 + kk16 + j;
                        kf[j] = qkv[(size_t)(b * S + key) * D3 + DMODEL + h * HDIM + hd];
                    }
                    if (half == 0)
                        s0 = __builtin_amdgcn_wmma_f32_16x16x32_bf16(false, qf[c], false, kf, (short)0, s0, false, false);
                    else
                        s1 = __builtin_amdgcn_wmma_f32_16x16x32_bf16(false, qf[c], false, kf, (short)0, s1, false, false);
                }
            }
        }
        // ---- scale, causal mask, online softmax ----
        int key0 = kc + (lane & 15);
#pragma unroll
        for (int r = 0; r < 8; ++r) {
            int qrow = qbase + r + hi8;
            float a0 = s0[r] * 0.125f;
            float a1 = s1[r] * 0.125f;
            if (key0 > qrow) a0 = -1e30f;
            if (key0 + 16 > qrow) a1 = -1e30f;
            float cm = fmaxf(a0, a1);
#pragma unroll
            for (int off = 1; off < 16; off <<= 1) cm = fmaxf(cm, __shfl_xor(cm, off, 32));
            float mn = fmaxf(mst[r], cm);
            float corr = __expf(mst[r] - mn);
            mst[r] = mn;
            float p0 = __expf(a0 - mn);
            float p1 = __expf(a1 - mn);
            float rs = p0 + p1;
#pragma unroll
            for (int off = 1; off < 16; off <<= 1) rs += __shfl_xor(rs, off, 32);
            lst[r] = lst[r] * corr + rs;
#pragma unroll
            for (int g = 0; g < 4; ++g) of[g][r] *= corr;
            int row = r + hi8;
            P[wid][row][lane & 15] = (bf16_t)p0;
            P[wid][row][16 + (lane & 15)] = (bf16_t)p1;
        }
        __syncthreads();
        // ---- P(16x32) @ V(32x64): A-frag from LDS, 4 B-frags from global ----
        v16bf pf;
#pragma unroll
        for (int j = 0; j < 16; ++j) {
            int kk = kb + j;
            pf[j] = P[wid][m][kk];
        }
        {
            int kk16 = (lane >> 4) * 16;
#pragma unroll
            for (int g = 0; g < 4; ++g) {
                v16bf vf;
#pragma unroll
                for (int j = 0; j < 16; ++j) {
                    int key = kc + kk16 + j;
                    int hd = g * 16 + (lane & 15);
                    vf[j] = qkv[(size_t)(b * S + key) * D3 + 2 * DMODEL + h * HDIM + hd];
                }
                of[g] = __builtin_amdgcn_wmma_f32_16x16x32_bf16(false, pf, false, vf, (short)0, of[g], false, false);
            }
        }
        __syncthreads();
    }

    // ---- normalize and store o as [token][D], col = h*64 + hd ----
#pragma unroll
    for (int r = 0; r < 8; ++r) {
        float inv = 1.f / lst[r];
        int row = qbase + r + hi8;
#pragma unroll
        for (int g = 0; g < 4; ++g) {
            int col = h * HDIM + g * 16 + (lane & 15);
            o[(size_t)(b * S + row) * DMODEL + col] = (bf16_t)(of[g][r] * inv);
        }
    }
}

// ---------------------------------------------------------------------------
// Router logits: logits[tok][e] = sum_d m[tok][d] * Wg[d][e]; 8 waves/token.
// ---------------------------------------------------------------------------
__global__ __launch_bounds__(256)
void logits_kernel(const bf16_t* __restrict__ mtok, const float* __restrict__ Wg,
                   float* __restrict__ logits) {
    int tok = blockIdx.x;
    int e = threadIdx.x >> 5;
    int lane = threadIdx.x & 31;
    float acc = 0.f;
    for (int d = lane; d < DMODEL; d += 32)
        acc += (float)mtok[(size_t)tok * DMODEL + d] * Wg[d * NEXP + e];
#pragma unroll
    for (int off = 16; off > 0; off >>= 1) acc += __shfl_down(acc, off, 32);
    if (lane == 0) logits[tok * NEXP + e] = acc;
}

__global__ void zero_counts_kernel(int* counts) {
    if (threadIdx.x < NEXP) counts[threadIdx.x] = 0;
}

// top-2 routing; renormalized weights = 2-way softmax over top-2 logits
__global__ __launch_bounds__(256)
void route_kernel(const float* __restrict__ logits, int* __restrict__ counts,
                  int* __restrict__ list, float* __restrict__ wts) {
    int tok = blockIdx.x * 256 + threadIdx.x;
    if (tok >= T_TOK) return;
    float l[NEXP];
#pragma unroll
    for (int e = 0; e < NEXP; ++e) l[e] = logits[tok * NEXP + e];
    int i1 = 0;
#pragma unroll
    for (int e = 1; e < NEXP; ++e) if (l[e] > l[i1]) i1 = e;
    int i2 = (i1 == 0) ? 1 : 0;
#pragma unroll
    for (int e = 0; e < NEXP; ++e) if (e != i1 && l[e] > l[i2]) i2 = e;
    float w1 = 1.f / (1.f + __expf(l[i2] - l[i1]));
    float w2 = 1.f - w1;
    int p1 = atomicAdd(&counts[i1], 1);
    list[i1 * T_TOK + p1] = tok; wts[i1 * T_TOK + p1] = w1;
    int p2 = atomicAdd(&counts[i2], 1);
    list[i2 * T_TOK + p2] = tok; wts[i2 * T_TOK + p2] = w2;
}

// ---------------------------------------------------------------------------
extern "C" void kernel_launch(void* const* d_in, const int* in_sizes, int n_in,
                              void* d_out, int out_size, void* d_ws, size_t ws_size,
                              hipStream_t stream) {
    const float* x    = (const float*)d_in[0];
    const float* Wqkv = (const float*)d_in[1];
    const float* Wo   = (const float*)d_in[2];
    const float* Wg   = (const float*)d_in[3];
    const float* W1   = (const float*)d_in[4];
    const float* W2   = (const float*)d_in[5];
    const float* g1   = (const float*)d_in[6];
    const float* b1   = (const float*)d_in[7];
    const float* g2   = (const float*)d_in[8];
    const float* b2   = (const float*)d_in[9];
    float* out = (float*)d_out;

    char* ws = (char*)d_ws;
    size_t off = 0;
    auto alloc = [&](size_t bytes) -> void* {
        void* p = ws + off;
        off += (bytes + 255) & ~(size_t)255;
        return p;
    };
    bf16_t* h_bf    = (bf16_t*)alloc((size_t)T_TOK * DMODEL * 2);
    bf16_t* qkv_bf  = (bf16_t*)alloc((size_t)T_TOK * 3 * DMODEL * 2);
    bf16_t* o_bf    = (bf16_t*)alloc((size_t)T_TOK * DMODEL * 2);
    float*  resid   = (float*)alloc((size_t)T_TOK * DMODEL * 4);
    bf16_t* m_bf    = (bf16_t*)alloc((size_t)T_TOK * DMODEL * 2);
    float*  logits  = (float*)alloc((size_t)T_TOK * NEXP * 4);
    int*    counts  = (int*)alloc(NEXP * 4);
    int*    list    = (int*)alloc((size_t)NEXP * T_TOK * 4);
    float*  wts     = (float*)alloc((size_t)NEXP * T_TOK * 4);
    bf16_t* he      = (bf16_t*)alloc((size_t)T_TOK * FFDIM * 2);
    (void)ws_size; (void)in_sizes; (void)n_in; (void)out_size;

    // 1) h = LN1(x) -> bf16
    ln_kernel<<<T_TOK, 256, 0, stream>>>(x, g1, b1, h_bf, nullptr);

    // 2) qkv = h @ Wqkv  (2048x3072x1024), bf16 out
    gemm_bf16<0, false><<<dim3(3 * DMODEL / 128, T_TOK / 64), 256, 0, stream>>>(
        h_bf, Wqkv, nullptr, qkv_bf, nullptr, nullptr, nullptr, nullptr,
        T_TOK, 3 * DMODEL, DMODEL);

    // 3) causal flash attention
    attn_kernel<<<dim3(SEQ / 128, NHEAD, NB), 256, 0, stream>>>(qkv_bf, o_bf);

    // 4) resid = o @ Wo + x  (fp32)
    gemm_bf16<1, false><<<dim3(DMODEL / 128, T_TOK / 64), 256, 0, stream>>>(
        o_bf, Wo, resid, nullptr, x, nullptr, nullptr, nullptr,
        T_TOK, DMODEL, DMODEL);

    // 5) m = LN2(resid) -> bf16; also seed d_out with resid (MoE accumulates on top)
    ln_kernel<<<T_TOK, 256, 0, stream>>>(resid, g2, b2, m_bf, out);

    // 6) router logits, top-2 routing with per-expert gather lists
    logits_kernel<<<T_TOK, 256, 0, stream>>>(m_bf, Wg, logits);
    zero_counts_kernel<<<1, 32, 0, stream>>>(counts);
    route_kernel<<<T_TOK / 256, 256, 0, stream>>>(logits, counts, list, wts);

    // 7) per expert: he = gelu(gather(m) @ W1[e]); out[tok] += w * (he @ W2[e])
    for (int e = 0; e < NEXP; ++e) {
        gemm_bf16<2, true><<<dim3(FFDIM / 128, T_TOK / 64), 256, 0, stream>>>(
            m_bf, W1 + (size_t)e * DMODEL * FFDIM, nullptr, he, nullptr,
            list + (size_t)e * T_TOK, nullptr, counts + e,
            T_TOK, FFDIM, DMODEL);
        gemm_bf16<3, false><<<dim3(DMODEL / 128, T_TOK / 64), 256, 0, stream>>>(
            he, W2 + (size_t)e * FFDIM * DMODEL, out, nullptr, nullptr,
            list + (size_t)e * T_TOK, wts + (size_t)e * T_TOK, counts + e,
            T_TOK, DMODEL, FFDIM);
    }
}
